// SiameseNetworkWithCrossAttention_55911884259682
// MI455X (gfx1250) — matrix operand order repaired and checked by reading, not compiled, not run on hardware
//
#include <hip/hip_runtime.h>
#include <hip/hip_bf16.h>

typedef _Float16 v16h __attribute__((ext_vector_type(16)));
typedef float    v8f  __attribute__((ext_vector_type(8)));
typedef unsigned int u32x4 __attribute__((ext_vector_type(4)));

union FragH { v16h v; u32x4 q[2]; };
union H8    { _Float16 h[8]; u32x4 q; };

__device__ __forceinline__ v8f vzero8() { v8f z = {0.f,0.f,0.f,0.f,0.f,0.f,0.f,0.f}; return z; }
__device__ __forceinline__ u32x4 qzero4() { u32x4 z = {0u,0u,0u,0u}; return z; }

__device__ __forceinline__ v8f wmma16(const FragH& a, const FragH& b, v8f c) {
  return __builtin_amdgcn_wmma_f32_16x16x32_f16(false, a.v, false, b.v, (short)0, c, false, false);
}

// A fragment: base -> k=0 of this row; hb = (lane&16)?8:0 ; k = hb+[0..8) , hb+16+[0..8)
__device__ __forceinline__ FragH load_fragA(const _Float16* base, int hb) {
  FragH f;
  f.q[0] = *(const u32x4*)(base + hb);
  f.q[1] = *(const u32x4*)(base + hb + 16);
  return f;
}
// B fragment: base -> k=0 of this column; kb = (lane&16)?16:0 ; k = kb/2+[0..8) , kb/2+16+[0..8)
__device__ __forceinline__ FragH load_fragB(const _Float16* base, int kb) {
  FragH f;
  f.q[0] = *(const u32x4*)(base + kb / 2);
  f.q[1] = *(const u32x4*)(base + kb / 2 + 16);
  return f;
}

// ---------------------------------------------------------------------------
// K0: repack weights to fp16, K-major.
//  w1h[64][32]  : k = (ky*3+kx)*3 + ic   (27 real, padded to 32)
//  w2h[128][576]: k = (ky*3+kx)*64 + ic
//  wqkv[160][128]: rows 0..15 wq, 16..31 wk, 32..159 wv
// ---------------------------------------------------------------------------
__global__ __launch_bounds__(256) void siam_prep(
    const float* __restrict__ c1w, const float* __restrict__ c2w,
    const float* __restrict__ wq, const float* __restrict__ wk, const float* __restrict__ wv,
    _Float16* __restrict__ w1h, _Float16* __restrict__ w2h, _Float16* __restrict__ wqkv) {
  int t = blockIdx.x * 256 + threadIdx.x;
  if (t < 2048) {
    int oc = t >> 5, kk = t & 31;
    float v = 0.f;
    if (kk < 27) { int pos = kk / 3, ic = kk - pos * 3; v = c1w[oc * 27 + ic * 9 + pos]; }
    w1h[t] = (_Float16)v;
  } else if (t < 2048 + 73728) {
    int i = t - 2048;
    int oc = i / 576, k = i - oc * 576;
    int pos = k >> 6, ic = k & 63;
    w2h[i] = (_Float16)c2w[oc * 576 + ic * 9 + pos];
  } else if (t < 2048 + 73728 + 20480) {
    int i = t - (2048 + 73728);
    int row = i >> 7, c = i & 127;
    float v;
    if (row < 16)       v = wq[row * 128 + c];
    else if (row < 32)  v = wk[(row - 16) * 128 + c];
    else                v = wv[(row - 32) * 128 + c];
    wqkv[i] = (_Float16)v;
  }
}

// ---------------------------------------------------------------------------
// K1: conv1(3x3,s2,p1,3->64) + bias + relu + maxpool2 -> f1p[ib][128][128][64] f16
// block: (pxb, py, ib=img*32+b), 256 thr. Computes conv rows 2py,2py+1 over 128 cx.
// ---------------------------------------------------------------------------
__global__ __launch_bounds__(256) void siam_conv1(
    const float* __restrict__ img1, const float* __restrict__ img2,
    const _Float16* __restrict__ w1h, const float* __restrict__ bias,
    _Float16* __restrict__ f1p) {
  __shared__ _Float16 col[256 * 40];      // im2col: 256 positions x 32 (pad 40)
  __shared__ _Float16 convbuf[256 * 72];  // relu'd conv tile: 256 pos x 64 oc (pad 72)
  const int tid = threadIdx.x;
  const int pxb = blockIdx.x, py = blockIdx.y, ib = blockIdx.z;
  const float* ibase = ((ib & 32) ? img2 : img1) + (size_t)(ib & 31) * (3 * 512 * 512);

  { // build im2col (k = pos*3+ic, zero pad/borders)
    int n = tid;
    int cy = 2 * py + (n >> 7);
    int cx = pxb * 128 + (n & 127);
    #pragma unroll
    for (int k = 0; k < 32; ++k) {
      float v = 0.f;
      if (k < 27) {
        int pos = k / 3, ic = k - pos * 3;
        int ky = pos / 3, kx = pos - ky * 3;
        int iy = 2 * cy - 1 + ky, ix = 2 * cx - 1 + kx;
        if ((unsigned)iy < 512u && (unsigned)ix < 512u)
          v = ibase[ic * 262144 + iy * 512 + ix];
      }
      col[n * 40 + k] = (_Float16)v;
    }
  }
  __syncthreads();

  const int lane = tid & 31, wid = tid >> 5;
  const int ln = lane & 15;
  const int hb = (lane & 16) ? 8 : 0;
  const int kb = (lane & 16) ? 16 : 0;
  const int mt = wid & 3, ntb = (wid >> 2) * 8;

  FragH a = load_fragA(w1h + (mt * 16 + ln) * 32, hb);
  #pragma unroll
  for (int j = 0; j < 8; ++j) {
    int nt = ntb + j;
    FragH bf = load_fragB(&col[(nt * 16 + ln) * 40], kb);
    v8f c = wmma16(a, bf, vzero8());
    H8 h8;
    #pragma unroll
    for (int r = 0; r < 8; ++r) {
      float vv = c[r] + bias[mt * 16 + hb + r];
      h8.h[r] = (_Float16)(vv > 0.f ? vv : 0.f);
    }
    *(u32x4*)&convbuf[(nt * 16 + ln) * 72 + mt * 16 + hb] = h8.q;
  }
  __syncthreads();

  // 2x2 maxpool -> one pooled row (py), 64 px, 64 oc
  int lx = tid >> 2, g = tid & 3;
  const _Float16* r0 = &convbuf[(2 * lx) * 72];
  const _Float16* r1 = &convbuf[(2 * lx + 1) * 72];
  const _Float16* r2 = &convbuf[(128 + 2 * lx) * 72];
  const _Float16* r3 = &convbuf[(128 + 2 * lx + 1) * 72];
  size_t ob = ((size_t)ib * 16384 + py * 128 + (pxb * 64 + lx)) * 64 + g * 16;
  #pragma unroll
  for (int gg = 0; gg < 2; ++gg) {
    H8 h8;
    #pragma unroll
    for (int r = 0; r < 8; ++r) {
      int oc = g * 16 + gg * 8 + r;
      _Float16 m0 = r0[oc] > r1[oc] ? r0[oc] : r1[oc];
      _Float16 m1 = r2[oc] > r3[oc] ? r2[oc] : r3[oc];
      h8.h[r] = m0 > m1 ? m0 : m1;
    }
    *(u32x4*)(f1p + ob + gg * 8) = h8.q;
  }
}

// ---------------------------------------------------------------------------
// K2: conv2(3x3,s2,p1,64->128) + bias + relu + maxpool2 -> f2p[ib][32][32][128] f16
// block: (py, ib), 256 thr / 8 waves. N=128 positions (2 conv rows x 64), K=576=18x32.
// Ping-pong im2col buffer: one barrier per K-chunk, next chunk's global loads
// issued before the current chunk's WMMAs so HBM/L2 latency overlaps the XDL pipe.
// ---------------------------------------------------------------------------
__global__ __launch_bounds__(256) void siam_conv2(
    const _Float16* __restrict__ f1p, const _Float16* __restrict__ w2h,
    const float* __restrict__ bias, _Float16* __restrict__ f2p) {
  __shared__ _Float16 colbuf[2][128 * 40]; // 128 pos x 32 k (pad 40), double buffered
  __shared__ _Float16 convbuf[128 * 136];  // 128 pos x 128 oc (pad 136)
  const int tid = threadIdx.x;
  const int py = blockIdx.x, ib = blockIdx.y;
  const _Float16* in = f1p + (size_t)ib * 16384 * 64;

  const int lane = tid & 31, wid = tid >> 5;
  const int ln = lane & 15;
  const int hb = (lane & 16) ? 8 : 0;
  const int kb = (lane & 16) ? 16 : 0;
  const int mt = wid;

  v8f acc[8];
  #pragma unroll
  for (int nt = 0; nt < 8; ++nt) acc[nt] = vzero8();

  const int n_ = tid >> 1, g_ = tid & 1;
  const int cy = 2 * py + (n_ >> 6), cx = n_ & 63;

  auto chunk_ptr = [&](int kc) -> const _Float16* {
    int pos = kc >> 1, icb = (kc & 1) * 32;
    int ky = pos / 3, kx = pos - ky * 3;
    int iy = 2 * cy - 1 + ky, ix = 2 * cx - 1 + kx;
    if ((unsigned)iy < 128u && (unsigned)ix < 128u)
      return in + ((size_t)iy * 128 + ix) * 64 + icb + g_ * 16;
    return nullptr;
  };
  auto load_chunk = [&](int kc, u32x4& d0, u32x4& d1) {
    const _Float16* sp = chunk_ptr(kc);
    d0 = qzero4(); d1 = qzero4();
    if (sp) { d0 = *(const u32x4*)sp; d1 = *(const u32x4*)(sp + 8); }
  };

  u32x4 d0, d1;
  load_chunk(0, d0, d1);
  int buf = 0;
  for (int kc = 0; kc < 18; ++kc) {
    _Float16* cb = colbuf[buf];
    *(u32x4*)&cb[n_ * 40 + g_ * 16] = d0;
    *(u32x4*)&cb[n_ * 40 + g_ * 16 + 8] = d1;
    __syncthreads();
    if (kc + 1 < 18) load_chunk(kc + 1, d0, d1);           // overlap with WMMAs below
    if (kc + 2 < 18) {                                     // L2 prefetch two chunks ahead
      const _Float16* pf = chunk_ptr(kc + 2);
      if (pf) __builtin_prefetch(pf, 0, 3);
    }
    FragH a = load_fragA(w2h + (mt * 16 + ln) * 576 + kc * 32, hb);
    #pragma unroll
    for (int nt = 0; nt < 8; ++nt) {
      FragH bf = load_fragB(&cb[(nt * 16 + ln) * 40], kb);
      acc[nt] = wmma16(a, bf, acc[nt]);
    }
    buf ^= 1;
  }

  #pragma unroll
  for (int nt = 0; nt < 8; ++nt) {
    H8 h8;
    #pragma unroll
    for (int r = 0; r < 8; ++r) {
      float vv = acc[nt][r] + bias[mt * 16 + hb + r];
      h8.h[r] = (_Float16)(vv > 0.f ? vv : 0.f);
    }
    *(u32x4*)&convbuf[(nt * 16 + ln) * 136 + mt * 16 + hb] = h8.q;
  }
  __syncthreads();

  int px = tid >> 3, gg = tid & 7, ocb = gg * 16;
  const _Float16* r0 = &convbuf[(2 * px) * 136];
  const _Float16* r1 = &convbuf[(2 * px + 1) * 136];
  const _Float16* r2 = &convbuf[(64 + 2 * px) * 136];
  const _Float16* r3 = &convbuf[(64 + 2 * px + 1) * 136];
  size_t ob = ((size_t)ib * 1024 + py * 32 + px) * 128 + ocb;
  #pragma unroll
  for (int h2 = 0; h2 < 2; ++h2) {
    H8 h8;
    #pragma unroll
    for (int r = 0; r < 8; ++r) {
      int oc = ocb + h2 * 8 + r;
      _Float16 m0 = r0[oc] > r1[oc] ? r0[oc] : r1[oc];
      _Float16 m1 = r2[oc] > r3[oc] ? r2[oc] : r3[oc];
      h8.h[r] = m0 > m1 ? m0 : m1;
    }
    *(u32x4*)(f2p + ob + h2 * 8) = h8.q;
  }
}

// ---------------------------------------------------------------------------
// K3: cross attention (row-wise, W=32) + residual + partial mean.
// block: (hgrp, b, dir), 128 thr = 4 waves, wave = one row h.
// pp[dir][b][h][128] = sum over w of (gamma*out + x1).
// ---------------------------------------------------------------------------
__global__ __launch_bounds__(128) void siam_attn(
    const _Float16* __restrict__ f2p, const _Float16* __restrict__ wqkv,
    const float* __restrict__ bq, const float* __restrict__ bk,
    const float* __restrict__ bv, const float* __restrict__ gptr,
    float* __restrict__ pp) {
  __shared__ _Float16 smem[4 * 7936];
  const int tid = threadIdx.x;
  const int lane = tid & 31, wid = tid >> 5;
  const int ln = lane & 15;
  const int hb = (lane & 16) ? 8 : 0;
  const int kb = (lane & 16) ? 16 : 0;
  const int hgrp = blockIdx.x, b = blockIdx.y, dir = blockIdx.z;
  const int h = hgrp * 4 + wid;
  const int i1 = dir ? (32 + b) : b;
  const int i2 = dir ? b : (32 + b);
  const _Float16* x1 = f2p + ((size_t)i1 * 1024 + h * 32) * 128;
  const _Float16* x2 = f2p + ((size_t)i2 * 1024 + h * 32) * 128;
  _Float16* q_l  = &smem[wid * 7936];   // qT[w][16]  stride 24
  _Float16* k_l  = q_l + 768;           // kT[w][16]  stride 24
  _Float16* at_l = q_l + 1536;          // attn[w][32] stride 40
  _Float16* vT_l = q_l + 2816;          // vT[c][32]  stride 40
  const float gamma = gptr[0];

  // qT = wq x1^T + bq ; kT = wk x2^T + bk   (16x32 each, K=128)
  #pragma unroll
  for (int which = 0; which < 2; ++which) {
    const _Float16* wrow = wqkv + which * 16 * 128 + ln * 128;
    const _Float16* xr = which ? x2 : x1;
    const float* bb = which ? bk : bq;
    _Float16* dst = which ? k_l : q_l;
    #pragma unroll
    for (int nt = 0; nt < 2; ++nt) {
      v8f c = vzero8();
      #pragma unroll
      for (int kc = 0; kc < 4; ++kc) {
        FragH a = load_fragA(wrow + kc * 32, hb);
        FragH bf = load_fragB(xr + (nt * 16 + ln) * 128 + kc * 32, kb);
        c = wmma16(a, bf, c);
      }
      H8 h8;
      #pragma unroll
      for (int r = 0; r < 8; ++r) h8.h[r] = (_Float16)(c[r] + bb[hb + r]);
      *(u32x4*)&dst[(nt * 16 + ln) * 24 + hb] = h8.q;
    }
  }

  // vT = wv x2^T + bv  (128x32, K=128)
  #pragma unroll
  for (int nt = 0; nt < 2; ++nt) {
    v8f vc[8];
    #pragma unroll
    for (int mt = 0; mt < 8; ++mt) vc[mt] = vzero8();
    #pragma unroll
    for (int kc = 0; kc < 4; ++kc) {
      FragH bf = load_fragB(x2 + (nt * 16 + ln) * 128 + kc * 32, kb);
      #pragma unroll
      for (int mt = 0; mt < 8; ++mt) {
        FragH a = load_fragA(wqkv + (32 + mt * 16 + ln) * 128 + kc * 32, hb);
        vc[mt] = wmma16(a, bf, vc[mt]);
      }
    }
    #pragma unroll
    for (int mt = 0; mt < 8; ++mt)
      #pragma unroll
      for (int r = 0; r < 8; ++r) {
        int cch = mt * 16 + hb + r;
        vT_l[cch * 40 + nt * 16 + ln] = (_Float16)(vc[mt][r] + bv[cch]);
      }
  }

  // scores s = q k^T (32x32), K=16 zero-padded to 32
  v8f s[2][2];
  #pragma unroll
  for (int mt = 0; mt < 2; ++mt) {
    FragH a;
    a.q[0] = *(const u32x4*)&q_l[(mt * 16 + ln) * 24 + hb];
    a.q[1] = qzero4();
    #pragma unroll
    for (int nt = 0; nt < 2; ++nt) {
      FragH bf;
      bf.q[0] = *(const u32x4*)&k_l[(nt * 16 + ln) * 24 + hb];
      bf.q[1] = qzero4();
      s[mt][nt] = wmma16(a, bf, vzero8());
    }
  }

  // row softmax (reduce across 16 lanes of each wave half)
  #pragma unroll
  for (int mt = 0; mt < 2; ++mt)
    #pragma unroll
    for (int r = 0; r < 8; ++r) {
      float v0 = s[mt][0][r], v1 = s[mt][1][r];
      float mx = fmaxf(v0, v1);
      #pragma unroll
      for (int msk = 1; msk < 16; msk <<= 1) mx = fmaxf(mx, __shfl_xor(mx, msk, 32));
      float e0 = __expf(v0 - mx), e1 = __expf(v1 - mx);
      float sm = e0 + e1;
      #pragma unroll
      for (int msk = 1; msk < 16; msk <<= 1) sm += __shfl_xor(sm, msk, 32);
      float inv = 1.f / sm;
      int w = mt * 16 + hb + r;
      at_l[w * 40 + ln]      = (_Float16)(e0 * inv);
      at_l[w * 40 + 16 + ln] = (_Float16)(e1 * inv);
    }

  // out = attn * v (32x128, K=32); residual + per-row channel sums
  #pragma unroll
  for (int nt = 0; nt < 8; ++nt) {
    float part = 0.f;
    #pragma unroll
    for (int mt = 0; mt < 2; ++mt) {
      FragH a = load_fragA(&at_l[(mt * 16 + ln) * 40], hb);
      FragH bf = load_fragB(&vT_l[(nt * 16 + ln) * 40], kb);
      v8f o = wmma16(a, bf, vzero8());
      #pragma unroll
      for (int r = 0; r < 8; ++r) {
        int w = mt * 16 + hb + r;
        part += gamma * o[r] + (float)x1[w * 128 + nt * 16 + ln];
      }
    }
    part += __shfl_xor(part, 16, 32);
    if (lane < 16)
      pp[(((size_t)(dir * 32 + b)) * 32 + h) * 128 + nt * 16 + ln] = part;
  }
}

// ---------------------------------------------------------------------------
// K4: reduce pp over h, /1024 (mean), apply fc1/fc2. block=(head,b), 256 thr.
// ---------------------------------------------------------------------------
__global__ __launch_bounds__(256) void siam_fc(
    const float* __restrict__ pp, const float* __restrict__ fc1w, const float* __restrict__ fc1b,
    const float* __restrict__ fc2w, const float* __restrict__ fc2b, float* __restrict__ out) {
  __shared__ float p[128];
  const int head = blockIdx.x, b = blockIdx.y, t = threadIdx.x;
  if (t < 128) {
    float s = 0.f;
    const float* base = pp + ((size_t)(head * 32 + b) * 32) * 128 + t;
    #pragma unroll
    for (int h2 = 0; h2 < 32; ++h2) s += base[h2 * 128];
    p[t] = s * (1.0f / 1024.0f);
  }
  __syncthreads();
  const float* W  = head ? fc2w : fc1w;
  const float* bi = head ? fc2b : fc1b;
  const float* wr = W + t * 128;
  float s = bi[t];
  #pragma unroll 8
  for (int c = 0; c < 128; ++c) s += p[c] * wr[c];
  out[head * 8192 + b * 256 + t] = s;
}

// ---------------------------------------------------------------------------
extern "C" void kernel_launch(void* const* d_in, const int* in_sizes, int n_in,
                              void* d_out, int out_size, void* d_ws, size_t ws_size,
                              hipStream_t stream) {
  (void)in_sizes; (void)n_in; (void)out_size; (void)ws_size;
  const float* img1 = (const float*)d_in[0];
  const float* img2 = (const float*)d_in[1];
  const float* c1w  = (const float*)d_in[2];
  const float* c1b  = (const float*)d_in[3];
  const float* c2w  = (const float*)d_in[4];
  const float* c2b  = (const float*)d_in[5];
  const float* wq   = (const float*)d_in[6];
  const float* bq   = (const float*)d_in[7];
  const float* wk   = (const float*)d_in[8];
  const float* bk   = (const float*)d_in[9];
  const float* wvp  = (const float*)d_in[10];
  const float* bv   = (const float*)d_in[11];
  const float* gam  = (const float*)d_in[12];
  const float* fc1w = (const float*)d_in[13];
  const float* fc1b = (const float*)d_in[14];
  const float* fc2w = (const float*)d_in[15];
  const float* fc2b = (const float*)d_in[16];

  char* ws = (char*)d_ws;
  size_t off = 0;
  auto take = [&](size_t bytes) { char* p = ws + off; off += (bytes + 255) & ~(size_t)255; return p; };
  _Float16* w1h   = (_Float16*)take((size_t)64 * 32 * 2);
  _Float16* w2h   = (_Float16*)take((size_t)128 * 576 * 2);
  _Float16* wqkvh = (_Float16*)take((size_t)160 * 128 * 2);
  float*    pp    = (float*)   take((size_t)2 * 32 * 32 * 128 * 4);
  _Float16* f1p   = (_Float16*)take((size_t)64 * 16384 * 64 * 2);   // 128 MB
  _Float16* f2p   = (_Float16*)take((size_t)64 * 1024 * 128 * 2);   // 16 MB

  siam_prep<<<376, 256, 0, stream>>>(c1w, c2w, wq, wk, wvp, w1h, w2h, wqkvh);
  siam_conv1<<<dim3(2, 128, 64), 256, 0, stream>>>(img1, img2, w1h, c1b, f1p);
  siam_conv2<<<dim3(32, 64), 256, 0, stream>>>(f1p, w2h, c2b, f2p);
  siam_attn<<<dim3(8, 32, 2), 128, 0, stream>>>(f2p, wqkvh, bq, bk, bv, gam, pp);
  siam_fc<<<dim3(2, 32), 256, 0, stream>>>(pp, fc1w, fc1b, fc2w, fc2b, (float*)d_out);
}
